// Attention_25649544691754
// MI455X (gfx1250) — matrix operand order repaired and checked by reading, not compiled, not run on hardware
//
#include <hip/hip_runtime.h>
#include <hip/hip_bf16.h>

// GQA paged-attention decode w/ fp8-e4m3 KV-cache round-trip, bf16 WMMA math,
// TDM (tensor_load_to_lds) staging of paged KV blocks with double buffering.
// 1 wave (32 lanes) per (sequence, kv-head); grid = 32*8 = 256 workgroups.

#define NHQ   32      // query heads
#define HD    128     // head dim
#define NKVH  8       // kv heads
#define GQ    4       // query heads per kv head
#define BSZ   16      // cache block size (== one 16-key WMMA tile)
#define MBLK  128     // max blocks per sequence
#define SEQS  32
#define LMAX  2048    // MBLK * BSZ
#define KSTR  132     // padded LDS row stride in floats (128 + 4) -> conflict-free

typedef __attribute__((ext_vector_type(16))) __bf16   v16bf;
typedef __attribute__((ext_vector_type(8)))  float    v8f;
typedef __attribute__((ext_vector_type(4)))  unsigned v4u;
typedef __attribute__((ext_vector_type(4)))  int      v4i;
typedef __attribute__((ext_vector_type(8)))  int      v8i_t;

#if defined(__has_builtin)
#if __has_builtin(__builtin_amdgcn_tensor_load_to_lds) && \
    __has_builtin(__builtin_amdgcn_s_wait_tensorcnt)
#define USE_TDM 1
#endif
#if __has_builtin(__builtin_amdgcn_cvt_pk_fp8_f32) && \
    __has_builtin(__builtin_amdgcn_cvt_pk_f32_fp8)
#define USE_HW_FP8 1
#endif
#endif
#ifndef USE_TDM
#define USE_TDM 0
#endif
#ifndef USE_HW_FP8
#define USE_HW_FP8 0
#endif

// f32 -> fp8 e4m3fn (RNE) -> f32, branchless software path.
// bias 7, max 448, subnormal quantum 2^-9. Idempotent on the fp8 grid.
__device__ __forceinline__ float fp8_e4m3_rt_sw(float x) {
  unsigned u  = __float_as_uint(x);
  unsigned au = u & 0x7fffffffu;
  float    ax = __uint_as_float(au);
  unsigned rn = (au + 0x7ffffu + ((au >> 20) & 1u)) & 0xfff00000u;
  float fn = fminf(__uint_as_float(rn), 448.0f);       // also saturates |x|>=448
  float fs = rintf(ax * 512.0f) * (1.0f / 512.0f);     // fp8-subnormal range
  float r  = (ax < 0.015625f) ? fs : fn;
  return copysignf(r, x);
}

// Round-trip two values through fp8-e4m3 (HW pk-convert when available).
__device__ __forceinline__ void fp8rt_pair(float x0, float x1, float& r0, float& r1) {
#if USE_HW_FP8
  int p = __builtin_amdgcn_cvt_pk_fp8_f32(x0, x1, 0, false);   // 2x fp8 in low word
  auto r = __builtin_amdgcn_cvt_pk_f32_fp8(p, false);          // decode back to f32
  r0 = r[0]; r1 = r[1];
#else
  r0 = fp8_e4m3_rt_sw(x0); r1 = fp8_e4m3_rt_sw(x1);
#endif
}

// 32-lane XOR-butterfly via ds_swizzle (group-of-32 mode: and=0x1f, or=0, xor=m).
#define SWZ_XOR(x, imm) \
  __int_as_float(__builtin_amdgcn_ds_swizzle(__float_as_int(x), imm))
__device__ __forceinline__ float wave_max(float v) {
  v = fmaxf(v, SWZ_XOR(v, 0x041f));   // ^1
  v = fmaxf(v, SWZ_XOR(v, 0x081f));   // ^2
  v = fmaxf(v, SWZ_XOR(v, 0x101f));   // ^4
  v = fmaxf(v, SWZ_XOR(v, 0x201f));   // ^8
  v = fmaxf(v, SWZ_XOR(v, 0x401f));   // ^16
  return v;
}
__device__ __forceinline__ float wave_sum(float v) {
  v += SWZ_XOR(v, 0x041f);
  v += SWZ_XOR(v, 0x081f);
  v += SWZ_XOR(v, 0x101f);
  v += SWZ_XOR(v, 0x201f);
  v += SWZ_XOR(v, 0x401f);
  return v;
}

#if USE_TDM
// DMA `rows` rows x 128 f32 (global row stride NKVH*HD f32) into LDS rows of
// KSTR f32 (TDM pad: 4 DWORDs after each 128 DWORDs -> pad_interval=6, amount=3).
__device__ __forceinline__ void stage_issue(float* lds_dst, const float* gsrc,
                                            int rows, int /*lane*/) {
  asm volatile("" ::: "memory");
  unsigned long long ga = (unsigned long long)(uintptr_t)gsrc;
  unsigned lds_off = (unsigned)(uintptr_t)(void*)lds_dst;  // low 32 bits = LDS offset
  v4u g0;
  g0[0] = 1u;                                    // count=1, user descriptor
  g0[1] = lds_off;                               // lds_addr
  g0[2] = (unsigned)ga;                          // global_addr[31:0]
  g0[3] = (unsigned)((ga >> 32) & 0x01ffffffu) | 0x80000000u;  // addr[56:32]|type=2
  v8i_t g1;
  g1[0] = (int)((2u << 16) | (1u << 20) | (6u << 22) | (3u << 25)); // 4B elems + pad
  g1[1] = (int)(128u << 16);                     // tensor_dim0 = 128
  g1[2] = (int)((unsigned)rows << 16);           // tensor_dim1 = rows
  g1[3] = (int)(128u << 16);                     // tile_dim0 = 128
  g1[4] = rows;                                  // tile_dim1 = rows, tile_dim2 = 0
  g1[5] = NKVH * HD;                             // tensor_dim0_stride = 1024 f32
  g1[6] = 0;
  g1[7] = 0;
  v4i z4 = {0, 0, 0, 0};
#if __clang_major__ >= 23
  v8i_t z8 = {0, 0, 0, 0, 0, 0, 0, 0};
  __builtin_amdgcn_tensor_load_to_lds(g0, g1, z4, z4, z8, 0);
#else
  __builtin_amdgcn_tensor_load_to_lds(g0, g1, z4, z4, 0);
#endif
}
__device__ __forceinline__ void stage_wait(int pending) {
  if (pending) __builtin_amdgcn_s_wait_tensorcnt(1);
  else         __builtin_amdgcn_s_wait_tensorcnt(0);
  asm volatile("" ::: "memory");
}
#else
// Fallback: cooperative wave copy global -> LDS (4 rows in flight, 8 lanes/row).
__device__ __forceinline__ void stage_issue(float* lds_dst, const float* gsrc,
                                            int rows, int lane) {
  for (int r = lane >> 3; r < rows; r += 4) {
    const float4* s = (const float4*)(gsrc + (size_t)r * (NKVH * HD));
    float4*       d = (float4*)(lds_dst + r * KSTR);
    int c = lane & 7;
    d[c] = s[c]; d[c + 8] = s[c + 8]; d[c + 16] = s[c + 16]; d[c + 24] = s[c + 24];
  }
}
__device__ __forceinline__ void stage_wait(int) { asm volatile("" ::: "memory"); }
#endif

__global__ __launch_bounds__(32) void paged_gqa_fp8_decode(
    const float* __restrict__ q,      const float* __restrict__ knew,
    const float* __restrict__ vnew,   const float* __restrict__ kcache,
    const float* __restrict__ vcache, const float* __restrict__ kscale,
    const float* __restrict__ vscale, const int* __restrict__ btab,
    const int* __restrict__ ctxlen,   float* __restrict__ out) {
  __shared__ float s_sc[GQ * LMAX];                         // scores, 32 KB
  __shared__ __align__(16) float s_stage[2 * BSZ * KSTR];   // KV stage, ~16.9 KB

  const int wg   = blockIdx.x;
  const int seq  = wg / NKVH;
  const int h    = wg % NKVH;
  const int lane = threadIdx.x;
  const int hl   = lane >> 4;          // which 16-lane half
  const int n16  = lane & 15;

  const int   ctx    = max(ctxlen[seq], 1);
  const int   ctxm1  = ctx - 1;
  const int   nt     = (ctx + BSZ - 1) / BSZ;   // 16-key tiles (one cache block each)
  const int   Lr     = nt * BSZ;
  const float kscl   = kscale[h], vscl = vscale[h];
  const float inv_k  = 1.0f / kscl, inv_v = 1.0f / vscl;
  const float SCL    = 0.08838834764831845f;    // 1/sqrt(128)
  const float NEGINF = -__builtin_inff();

  // ---- Q -> WMMA A fragments (16 rows: 4 valid query heads + 12 zero rows), bf16.
  // A layout: lane(hl,n16) = row n16; elem j=2v+p -> d = 32c + 8*hl + 2*(v&3)+p + (v>=4?16:0)
  v16bf aq[4];
  for (int c = 0; c < 4; ++c)
    for (int j = 0; j < 16; ++j) {
      int vg = j >> 1, pp = j & 1;
      int dof = c * 32 + hl * 8 + ((vg & 3) << 1) + pp + ((vg >= 4) ? 16 : 0);
      float x = (n16 < GQ)
          ? q[(size_t)seq * NHQ * HD + (size_t)(h * GQ + n16) * HD + dof] : 0.0f;
      aq[c][j] = (__bf16)x;
    }

  // ---- Phase 1: scores for every 16-key tile -> LDS, TDM double-buffered K.
  {
    int bt0 = btab[seq * MBLK];
    stage_issue(s_stage, kcache + ((size_t)(bt0 * BSZ) * NKVH + h) * HD, BSZ, lane);
  }
  for (int t = 0; t < nt; ++t) {
    float* cur = s_stage + (t & 1) * (BSZ * KSTR);
    if (t + 1 < nt) {
      int btn = btab[seq * MBLK + t + 1];
      stage_issue(s_stage + ((t + 1) & 1) * (BSZ * KSTR),
                  kcache + ((size_t)(btn * BSZ) * NKVH + h) * HD, BSZ, lane);
      stage_wait(1);                 // tile t landed; t+1 still in flight
    } else {
      stage_wait(0);
    }
    if (t == (ctxm1 >> 4)) {         // new-token fixup: store fp8-quantized k/scale
      int d0 = lane * 4;
      float4 kx = *(const float4*)(knew + (size_t)seq * NKVH * HD + (size_t)h * HD + d0);
      float r0, r1, r2, r3;
      fp8rt_pair(kx.x * inv_k, kx.y * inv_k, r0, r1);
      fp8rt_pair(kx.z * inv_k, kx.w * inv_k, r2, r3);
      *(float4*)&cur[(ctxm1 & 15) * KSTR + d0] = make_float4(r0, r1, r2, r3);
    }
    v8f acc = {};
    for (int c = 0; c < 4; ++c) {
      // B layout: lane(hl,n16) = key column n16, elem j -> d = 32c + 8*hl + j + (j>=8?8:0)
      const float* rowp = cur + n16 * KSTR + c * 32 + hl * 8;
      v16bf bk;
      for (int j = 0; j < 16; j += 2) {
        float r0, r1;
        fp8rt_pair(rowp[j + ((j >= 8) ? 8 : 0)], rowp[j + 1 + ((j >= 8) ? 8 : 0)], r0, r1);
        bk[j]     = (__bf16)(r0 * kscl);
        bk[j + 1] = (__bf16)(r1 * kscl);
      }
      acc = __builtin_amdgcn_wmma_f32_16x16x32_bf16(false, aq[c], false, bk,
                                                    (short)0, acc, false, false);
    }
    if (hl == 0) {                   // D: VGPR g = query-head row, col = key n16
      int kidx = t * BSZ + n16;
      for (int g = 0; g < GQ; ++g)
        s_sc[g * LMAX + kidx] = (kidx < ctx) ? acc[g] * SCL : NEGINF;
    }
  }
  __syncthreads();

  // ---- Phase 2: row max & sum(exp); float4 LDS scans + ds_swizzle butterflies.
  float m[GQ], lsum[GQ];
  for (int g = 0; g < GQ; ++g) {
    float mm = NEGINF;
    for (int l0 = lane * 4; l0 < Lr; l0 += 128) {
      float4 s4 = *(const float4*)&s_sc[g * LMAX + l0];
      mm = fmaxf(fmaxf(fmaxf(mm, s4.x), s4.y), fmaxf(s4.z, s4.w));
    }
    mm = wave_max(mm);
    float ss = 0.0f;
    for (int l0 = lane * 4; l0 < Lr; l0 += 128) {
      float4 s4 = *(const float4*)&s_sc[g * LMAX + l0];
      ss += __expf(s4.x - mm) + __expf(s4.y - mm) +
            __expf(s4.z - mm) + __expf(s4.w - mm);
    }
    ss = wave_sum(ss);
    m[g] = mm; lsum[g] = ss;
  }

  // ---- Phase 3: O += P * V over 32-key windows (two TDM-staged cache blocks).
  v8f zero = {};
  v8f oacc[8];
  for (int j = 0; j < 8; ++j) oacc[j] = zero;
  const int nt2 = (nt + 1) >> 1;
  for (int t2 = 0; t2 < nt2; ++t2) {
    const int kb = t2 * 32;
    {
      int ba = btab[seq * MBLK + 2 * t2];
      stage_issue(s_stage, vcache + ((size_t)(ba * BSZ) * NKVH + h) * HD, BSZ, lane);
      int ib = 2 * t2 + 1;
      int bb = btab[seq * MBLK + ((ib < MBLK) ? ib : (MBLK - 1))];
      stage_issue(s_stage + BSZ * KSTR,
                  vcache + ((size_t)(bb * BSZ) * NKVH + h) * HD, BSZ, lane);
      stage_wait(0);
    }
    if (t2 == (ctxm1 >> 5)) {        // new-token fixup for V
      int d0 = lane * 4;
      float4 vx = *(const float4*)(vnew + (size_t)seq * NKVH * HD + (size_t)h * HD + d0);
      float r0, r1, r2, r3;
      fp8rt_pair(vx.x * inv_v, vx.y * inv_v, r0, r1);
      fp8rt_pair(vx.z * inv_v, vx.w * inv_v, r2, r3);
      *(float4*)&s_stage[(ctxm1 & 31) * KSTR + d0] = make_float4(r0, r1, r2, r3);
    }
    // P (A operand): row = query head n16 (<4), elem j -> key kb + 8*hl + 2*(v&3)+p + (v>=4?16:0)
    v16bf pf;
    for (int j = 0; j < 16; ++j) {
      int vg = j >> 1, pp = j & 1;
      int ko = hl * 8 + ((vg & 3) << 1) + pp + ((vg >= 4) ? 16 : 0);
      int kidx = kb + ko;
      float pv = 0.0f;
      if (n16 < GQ && kidx < Lr) pv = __expf(s_sc[n16 * LMAX + kidx] - m[n16]);
      pf[j] = (__bf16)pv;
    }
    // V (B operand): col d = 16j + n16, elem e -> key 8*hl + e + (e>=8?8:0); dequant fused
    for (int j = 0; j < 8; ++j) {
      const int dcol = j * 16 + n16;
      v16bf bv;
      for (int e = 0; e < 16; e += 2) {
        float r0, r1;
        fp8rt_pair(s_stage[(hl * 8 + e + ((e >= 8) ? 8 : 0)) * KSTR + dcol],
                   s_stage[(hl * 8 + e + 1 + ((e >= 8) ? 8 : 0)) * KSTR + dcol], r0, r1);
        bv[e]     = (__bf16)(r0 * vscl);
        bv[e + 1] = (__bf16)(r1 * vscl);
      }
      oacc[j] = __builtin_amdgcn_wmma_f32_16x16x32_bf16(false, pf, false, bv,
                                                        (short)0, oacc[j], false, false);
    }
  }

  // ---- Epilogue: divide by softmax sum, write o[seq, h*4+g, d].
  if (hl == 0) {
    for (int j = 0; j < 8; ++j)
      for (int g = 0; g < GQ; ++g)
        out[(size_t)seq * NHQ * HD + (size_t)(h * GQ + g) * HD + j * 16 + n16] =
            oacc[j][g] / lsum[g];
  }
}

extern "C" void kernel_launch(void* const* d_in, const int* in_sizes, int n_in,
                              void* d_out, int out_size, void* d_ws, size_t ws_size,
                              hipStream_t stream) {
  (void)in_sizes; (void)n_in; (void)out_size; (void)d_ws; (void)ws_size;
  const float* q      = (const float*)d_in[0];
  const float* knew   = (const float*)d_in[1];
  const float* vnew   = (const float*)d_in[2];
  const float* kcache = (const float*)d_in[3];
  const float* vcache = (const float*)d_in[4];
  const float* kscale = (const float*)d_in[5];
  const float* vscale = (const float*)d_in[6];
  // d_in[7] = slot_mapping: derivable from block_tables + context_lens; unused.
  const int*   btab   = (const int*)d_in[8];
  const int*   ctxlen = (const int*)d_in[9];
  float* out = (float*)d_out;

  dim3 grid(SEQS * NKVH), block(32);
  hipLaunchKernelGGL(paged_gqa_fp8_decode, grid, block, 0, stream,
                     q, knew, vnew, kcache, vcache, kscale, vscale, btab, ctxlen, out);
}
